// SS2Dv2_33432025432195
// MI455X (gfx1250) — matrix operand order repaired and checked by reading, not compiled, not run on hardware
//
#include <hip/hip_runtime.h>

// ---------------------------------------------------------------------------
// SS2D (4-direction Mamba2/SSD) forward for MI455X (gfx1250), bf16 WMMA path.
// ---------------------------------------------------------------------------

typedef __bf16 bf16;
typedef __attribute__((ext_vector_type(16))) __bf16 v16bf;
typedef __attribute__((ext_vector_type(8)))  __bf16 v8bf;
typedef __attribute__((ext_vector_type(8)))  float  v8f;

#define BATCH   4
#define HIMG    64
#define WIMG    64
#define LSEQ    4096      // HIMG*WIMG
#define DMODEL  192
#define DINNER  384
#define NH3     3         // heads per direction
#define HT      12        // total heads (4 dirs * 3)
#define PDIM    128       // head dim
#define NSTATE  256       // 4 * D_STATE
#define CONVD   512
#define DIP     899       // in-proj output width
#define CHUNK   256
#define NCHUNK  16

__device__ __forceinline__ v8f wmma_bf16(v16bf a, v16bf b, v8f c) {
  return __builtin_amdgcn_wmma_f32_16x16x32_bf16(
      /*neg_a=*/false, a, /*neg_b=*/false, b,
      /*c_mod=*/(short)0, c, /*reuse_a=*/false, /*reuse_b=*/false);
}

// A-fragment K index for element e2 (contiguous k = 16*kh + e2, as ldfragA/ldfragB load)
__device__ __forceinline__ int a_kof(int kh, int e2) {
  return kh * 16 + e2;
}

// A-fragment: row-major source, one contiguous 16-elem run at kh*16
__device__ __forceinline__ v16bf ldfragA(const bf16* row, int kh) {
  v8bf lo = *(const v8bf*)(row + kh * 16);
  v8bf hi = *(const v8bf*)(row + kh * 16 + 8);
  return __builtin_shufflevector(lo, hi, 0, 1, 2, 3, 4, 5, 6, 7,
                                 8, 9, 10, 11, 12, 13, 14, 15);
}
// B-fragment: one contiguous 16-elem run at kh*16
__device__ __forceinline__ v16bf ldfragB(const bf16* row, int kh) {
  v8bf lo = *(const v8bf*)(row + kh * 16);
  v8bf hi = *(const v8bf*)(row + kh * 16 + 8);
  return __builtin_shufflevector(lo, hi, 0, 1, 2, 3, 4, 5, 6, 7,
                                 8, 9, 10, 11, 12, 13, 14, 15);
}

// scan index t -> spatial location for direction d (H = W = 64)
__device__ __forceinline__ int spatial_of(int d, int t) {
  switch (d) {
    case 0:  return t;
    case 1:  return ((t & 63) << 6) | (t >> 6);
    case 2:  return 4095 - t;
    default: { int u = 4095 - t; return ((u & 63) << 6) | (u >> 6); }
  }
}

__device__ __forceinline__ float sigmoidf_(float x) {
  return 1.0f / (1.0f + __expf(-x));
}

// ---------------------------------------------------------------------------
// GEMM: C[M,N] = A[M,K] * W[N,K]^T  (bf16 WMMA, LDS staged, compile-time dims)
// Requires M%64==0, K%32==0.  N may have a tail (clamped loads + select).
// grid: (ceil(N/64), M/64), block 256 (8 waves, 2 subtiles each)
// ---------------------------------------------------------------------------
template <int M, int N, int K>
__global__ void k_gemm(const float* __restrict__ A, const float* __restrict__ W,
                       float* __restrict__ C) {
  constexpr bool NTAIL = (N % 64) != 0;
  __shared__ bf16 As[64][40];
  __shared__ bf16 Bs[64][40];
  const int tid  = threadIdx.x;
  const int lane = tid & 31, wave = tid >> 5;
  const int m0 = blockIdx.y * 64, n0 = blockIdx.x * 64;
  const int mi = wave >> 2, ni = wave & 3;
  const int lr = lane & 15, kh = lane >> 4;
  const int r = tid >> 2, kc = (tid & 3) * 8;

  const int gn = n0 + r;
  const int gnc = NTAIL ? (gn < N ? gn : N - 1) : gn;
  const float* arow = A + (size_t)(m0 + r) * K;
  const float* wrow = W + (size_t)gnc * K;
  v8f acc0 = {}; v8f acc1 = {};

#pragma unroll
  for (int k0 = 0; k0 < K; k0 += 32) {
    if (k0 + 32 < K) {
      __builtin_prefetch(arow + k0 + 32 + kc, 0, 0);   // global_prefetch_b8
      __builtin_prefetch(wrow + k0 + 32 + kc, 0, 0);
    }
    float4 a01 = *(const float4*)(arow + k0 + kc);
    float4 a23 = *(const float4*)(arow + k0 + kc + 4);
    float4 w01 = *(const float4*)(wrow + k0 + kc);
    float4 w23 = *(const float4*)(wrow + k0 + kc + 4);
    if (NTAIL && gn >= N) {
      w01 = make_float4(0.f, 0.f, 0.f, 0.f);
      w23 = make_float4(0.f, 0.f, 0.f, 0.f);
    }
    v8bf pa, pw;
    pa[0] = (bf16)a01.x; pa[1] = (bf16)a01.y; pa[2] = (bf16)a01.z; pa[3] = (bf16)a01.w;
    pa[4] = (bf16)a23.x; pa[5] = (bf16)a23.y; pa[6] = (bf16)a23.z; pa[7] = (bf16)a23.w;
    pw[0] = (bf16)w01.x; pw[1] = (bf16)w01.y; pw[2] = (bf16)w01.z; pw[3] = (bf16)w01.w;
    pw[4] = (bf16)w23.x; pw[5] = (bf16)w23.y; pw[6] = (bf16)w23.z; pw[7] = (bf16)w23.w;
    *(v8bf*)(&As[r][kc]) = pa;
    *(v8bf*)(&Bs[r][kc]) = pw;
    __syncthreads();

    v16bf a0 = ldfragA(&As[mi * 16 + lr][0], kh);
    v16bf a1 = ldfragA(&As[(mi + 2) * 16 + lr][0], kh);
    v16bf bf = ldfragB(&Bs[ni * 16 + lr][0], kh);
    acc0 = wmma_bf16(a0, bf, acc0);
    acc1 = wmma_bf16(a1, bf, acc1);
    __syncthreads();
  }
  const int col = n0 + ni * 16 + lr;
  if (!NTAIL || col < N) {
#pragma unroll
    for (int rr = 0; rr < 8; ++rr) {
      int row0 = m0 + mi * 16 + kh * 8 + rr;
      C[(size_t)row0 * N + col]        = acc0[rr];
      C[(size_t)(row0 + 32) * N + col] = acc1[rr];
    }
  }
}

// ---------------------------------------------------------------------------
// dt softplus: dts[b,l,3] = softplus(zxbcdt[...,896+h3] + dt_bias[h3])
// ---------------------------------------------------------------------------
__global__ void k_dts(const float* __restrict__ zxbcdt, const float* __restrict__ dt_bias,
                      float* __restrict__ dts) {
  int idx = blockIdx.x * blockDim.x + threadIdx.x;   // over B*L
  if (idx >= BATCH * LSEQ) return;
#pragma unroll
  for (int h3 = 0; h3 < NH3; ++h3) {
    float x = zxbcdt[(size_t)idx * DIP + DINNER + CONVD + h3] + dt_bias[h3];
    dts[(size_t)idx * NH3 + h3] = (x > 20.0f) ? x : log1pf(__expf(x));
  }
}

// ---------------------------------------------------------------------------
// Depthwise 3x3 conv (pad 1) + bias + SiLU.  out convo[b][c][loc]
// ---------------------------------------------------------------------------
__global__ void k_conv(const float* __restrict__ zxbcdt, const float* __restrict__ cw,
                       const float* __restrict__ cb, float* __restrict__ convo) {
  int idx = blockIdx.x * blockDim.x + threadIdx.x;   // B*CONVD*LSEQ
  int loc = idx & (LSEQ - 1);
  int c   = (idx >> 12) & (CONVD - 1);
  int b   = idx >> 21;
  int hh = loc >> 6, ww = loc & 63;
  float s = cb[c];
#pragma unroll
  for (int ky = 0; ky < 3; ++ky) {
#pragma unroll
    for (int kx = 0; kx < 3; ++kx) {
      int y2 = hh + ky - 1, x2 = ww + kx - 1;
      if (y2 >= 0 && y2 < HIMG && x2 >= 0 && x2 < WIMG) {
        s += cw[c * 9 + ky * 3 + kx] *
             zxbcdt[((size_t)b * LSEQ + (y2 << 6) + x2) * DIP + DINNER + c];
      }
    }
  }
  s = s * sigmoidf_(s);
  convo[((size_t)b * CONVD + c) * LSEQ + loc] = s;
}

// ---------------------------------------------------------------------------
// Per-chunk inclusive cumsum of dA = dts * (-exp(A_log)).  192 lanes total.
// ---------------------------------------------------------------------------
__global__ void k_cumsum(const float* __restrict__ dts, const float* __restrict__ A_log,
                         float* __restrict__ Acum, float* __restrict__ Alast) {
  int idx = blockIdx.x * blockDim.x + threadIdx.x;   // B*3*NCHUNK = 192
  if (idx >= BATCH * NH3 * NCHUNK) return;
  int c  = idx & (NCHUNK - 1);
  int h3 = (idx >> 4) % NH3;
  int b  = idx / (NH3 * NCHUNK);
  float a = -__expf(A_log[h3]);
  float run = 0.0f;
  for (int i = 0; i < CHUNK; ++i) {
    int t = c * CHUNK + i;
    run += a * dts[((size_t)b * LSEQ + t) * NH3 + h3];
    Acum[((size_t)b * NH3 + h3) * LSEQ + t] = run;
  }
  Alast[(b * NH3 + h3) * NCHUNK + c] = run;
}

// ---------------------------------------------------------------------------
// Gather B/C (concat of 4 directions) -> Bm[b][t][n], BmT[b][n][t], Cm[b][t][n]
// ---------------------------------------------------------------------------
__global__ void k_gatherBC(const float* __restrict__ convo, bf16* __restrict__ Bm,
                           bf16* __restrict__ BmT, bf16* __restrict__ Cm) {
  int idx = blockIdx.x * blockDim.x + threadIdx.x;   // B*L*256
  int n = idx & (NSTATE - 1);
  int t = (idx >> 8) & (LSEQ - 1);
  int b = idx >> 20;
  int d = n >> 6, nn = n & 63;
  int loc = spatial_of(d, t);
  float bv = convo[((size_t)b * CONVD + DINNER + nn) * LSEQ + loc];
  float cv = convo[((size_t)b * CONVD + DINNER + 64 + nn) * LSEQ + loc];
  Bm[idx] = (bf16)bv;
  Cm[idx] = (bf16)cv;
  BmT[((size_t)b * NSTATE + n) * LSEQ + t] = (bf16)bv;
}

// ---------------------------------------------------------------------------
// Gather dt-scaled X transposed: XdT[b][h][p][t]
// ---------------------------------------------------------------------------
__global__ void k_gatherX(const float* __restrict__ convo, const float* __restrict__ dts,
                          bf16* __restrict__ XdT) {
  int idx = blockIdx.x * blockDim.x + threadIdx.x;   // B*HT*P*L
  int t = idx & (LSEQ - 1);
  int p = (idx >> 12) & (PDIM - 1);
  int rest = idx >> 19;
  int h = rest % HT;
  int b = rest / HT;
  int d = h / NH3, h3 = h % NH3;
  int loc = spatial_of(d, t);
  float v = convo[((size_t)b * CONVD + h3 * PDIM + p) * LSEQ + loc] *
            dts[((size_t)b * LSEQ + t) * NH3 + h3];
  XdT[idx] = (bf16)v;
}

// ---------------------------------------------------------------------------
// CB[b,c,l,s] = sum_n C[l,n]*B[s,n]   (head-independent).  256x256x256 WMMA.
// grid (4,4,B*NCHUNK)
// ---------------------------------------------------------------------------
__global__ void k_cb(const bf16* __restrict__ Cm, const bf16* __restrict__ Bm,
                     bf16* __restrict__ CB) {
  int z = blockIdx.z, b = z >> 4, c = z & 15;
  const int lane = threadIdx.x & 31, wave = threadIdx.x >> 5;
  const int m0 = blockIdx.y * 64, n0 = blockIdx.x * 64;
  const int mi = wave >> 2, ni = wave & 3;
  const int lr = lane & 15, kh = lane >> 4;
  const bf16* crow0 = Cm + ((size_t)b * LSEQ + c * CHUNK + m0 + mi * 16 + lr) * NSTATE;
  const bf16* crow1 = crow0 + 32 * NSTATE;
  const bf16* brow  = Bm + ((size_t)b * LSEQ + c * CHUNK + n0 + ni * 16 + lr) * NSTATE;
  v8f acc0 = {}; v8f acc1 = {};
#pragma unroll
  for (int k0 = 0; k0 < NSTATE; k0 += 32) {
    v16bf a0 = ldfragA(crow0 + k0, kh);
    v16bf a1 = ldfragA(crow1 + k0, kh);
    v16bf bf = ldfragB(brow + k0, kh);
    acc0 = wmma_bf16(a0, bf, acc0);
    acc1 = wmma_bf16(a1, bf, acc1);
  }
  bf16* out = CB + (size_t)z * CHUNK * CHUNK;
  int col = n0 + ni * 16 + lr;
#pragma unroll
  for (int rr = 0; rr < 8; ++rr) {
    int row0 = m0 + mi * 16 + kh * 8 + rr;
    out[(size_t)row0 * CHUNK + col]        = (bf16)acc0[rr];
    out[(size_t)(row0 + 32) * CHUNK + col] = (bf16)acc1[rr];
  }
}

// ---------------------------------------------------------------------------
// states[b,c,h,p,n] = sum_l XdT[p,l]*exp(Alast-Acum[l]) * BmT[n,l]
// grid (4,2,B*NCHUNK*HT)
// ---------------------------------------------------------------------------
__global__ void k_states(const bf16* __restrict__ XdT, const bf16* __restrict__ BmT,
                         const float* __restrict__ Acum, const float* __restrict__ Alast,
                         float* __restrict__ states) {
  __shared__ float decay[CHUNK];
  int z = blockIdx.z;
  int h = z % HT; int bc = z / HT; int b = bc >> 4; int c = bc & 15;
  int h3 = h % NH3;
  const float* ac = Acum + ((size_t)b * NH3 + h3) * LSEQ + c * CHUNK;
  float alast = Alast[(b * NH3 + h3) * NCHUNK + c];
  decay[threadIdx.x & (CHUNK - 1)] = __expf(alast - ac[threadIdx.x & (CHUNK - 1)]);
  __syncthreads();

  const int lane = threadIdx.x & 31, wave = threadIdx.x >> 5;
  const int m0 = blockIdx.y * 64, n0 = blockIdx.x * 64;   // m over p, n over states
  const int mi = wave >> 2, ni = wave & 3;
  const int lr = lane & 15, kh = lane >> 4;
  const int p0 = m0 + mi * 16 + lr, p1 = p0 + 32;
  const int ncol = n0 + ni * 16 + lr;
  const bf16* xrow0 = XdT + ((size_t)(b * HT + h) * PDIM + p0) * LSEQ + c * CHUNK;
  const bf16* xrow1 = XdT + ((size_t)(b * HT + h) * PDIM + p1) * LSEQ + c * CHUNK;
  const bf16* btrow = BmT + ((size_t)b * NSTATE + ncol) * LSEQ + c * CHUNK;
  v8f acc0 = {}; v8f acc1 = {};
#pragma unroll
  for (int k0 = 0; k0 < CHUNK; k0 += 32) {
    v16bf x0 = ldfragA(xrow0 + k0, kh);
    v16bf x1 = ldfragA(xrow1 + k0, kh);
    v16bf a0, a1;
#pragma unroll
    for (int i = 0; i < 16; ++i) {
      float dcy = decay[k0 + a_kof(kh, i)];
      a0[i] = (bf16)((float)x0[i] * dcy);
      a1[i] = (bf16)((float)x1[i] * dcy);
    }
    v16bf bf = ldfragB(btrow + k0, kh);
    acc0 = wmma_bf16(a0, bf, acc0);
    acc1 = wmma_bf16(a1, bf, acc1);
  }
  float* out = states + (size_t)z * PDIM * NSTATE;
#pragma unroll
  for (int rr = 0; rr < 8; ++rr) {
    int row0 = m0 + mi * 16 + kh * 8 + rr;
    out[(size_t)row0 * NSTATE + ncol]        = acc0[rr];
    out[(size_t)(row0 + 32) * NSTATE + ncol] = acc1[rr];
  }
}

// ---------------------------------------------------------------------------
// Inter-chunk recurrence (in place): states[c] becomes prev-state for chunk c.
// ---------------------------------------------------------------------------
__global__ void k_recur(float* __restrict__ states, const float* __restrict__ Alast) {
  int idx = blockIdx.x * blockDim.x + threadIdx.x;   // B*HT*P*NSTATE
  int n = idx & (NSTATE - 1);
  int p = (idx >> 8) & (PDIM - 1);
  int rest = idx >> 15;
  int h = rest % HT;
  int b = rest / HT;
  int h3 = h % NH3;
  float run = 0.0f;
  for (int c = 0; c < NCHUNK; ++c) {
    size_t si = ((((size_t)b * NCHUNK + c) * HT + h) * PDIM + p) * NSTATE + n;
    float s = states[si];
    states[si] = run;
    run = run * __expf(Alast[(b * NH3 + h3) * NCHUNK + c]) + s;
  }
}

// ---------------------------------------------------------------------------
// Y[l,p] = sum_{s<=l} CB[l,s]*exp(Acum[l]-Acum[s]) * XdT[p,s]
//        + sum_n Cm[l,n]*exp(Acum[l]) * prev[p,n]
// grid (2,4,B*NCHUNK*HT)
// ---------------------------------------------------------------------------
__global__ void k_y(const bf16* __restrict__ CB, const bf16* __restrict__ XdT,
                    const bf16* __restrict__ Cm, const float* __restrict__ prev,
                    const float* __restrict__ Acum, float* __restrict__ Y) {
  __shared__ float acs[CHUNK];
  int z = blockIdx.z;
  int h = z % HT; int bc = z / HT; int b = bc >> 4; int c = bc & 15;
  int h3 = h % NH3;
  const float* ac = Acum + ((size_t)b * NH3 + h3) * LSEQ + c * CHUNK;
  acs[threadIdx.x & (CHUNK - 1)] = ac[threadIdx.x & (CHUNK - 1)];
  __syncthreads();

  const int lane = threadIdx.x & 31, wave = threadIdx.x >> 5;
  const int m0 = blockIdx.y * 64;   // l tile
  const int p0 = blockIdx.x * 64;   // p tile
  const int mi = wave >> 2, ni = wave & 3;
  const int lr = lane & 15, kh = lane >> 4;
  const int l0 = m0 + mi * 16 + lr, l1 = l0 + 32;
  const int pcol = p0 + ni * 16 + lr;
  const float ac0 = acs[l0], ac1 = acs[l1];
  const bf16* cbrow0 = CB + ((size_t)(b * NCHUNK + c)) * CHUNK * CHUNK + (size_t)l0 * CHUNK;
  const bf16* cbrow1 = cbrow0 + 32 * CHUNK;
  const bf16* xprow = XdT + ((size_t)(b * HT + h) * PDIM + pcol) * LSEQ + c * CHUNK;
  const bf16* cm0 = Cm + ((size_t)b * LSEQ + c * CHUNK + l0) * NSTATE;
  const bf16* cm1 = cm0 + 32 * NSTATE;
  const float* pvrow = prev + (((size_t)(b * NCHUNK + c) * HT + h)) * PDIM * NSTATE +
                       (size_t)pcol * NSTATE;
  v8f acc0 = {}; v8f acc1 = {};

  // phase 1: causal-masked (L * CB) @ Xd
#pragma unroll
  for (int k0 = 0; k0 < CHUNK; k0 += 32) {
    v16bf c0 = ldfragA(cbrow0 + k0, kh);
    v16bf c1 = ldfragA(cbrow1 + k0, kh);
    v16bf a0, a1;
#pragma unroll
    for (int i = 0; i < 16; ++i) {
      int s = k0 + a_kof(kh, i);
      float acS = acs[s];
      a0[i] = (s <= l0) ? (bf16)((float)c0[i] * __expf(ac0 - acS)) : (bf16)0.0f;
      a1[i] = (s <= l1) ? (bf16)((float)c1[i] * __expf(ac1 - acS)) : (bf16)0.0f;
    }
    v16bf bf = ldfragB(xprow + k0, kh);
    acc0 = wmma_bf16(a0, bf, acc0);
    acc1 = wmma_bf16(a1, bf, acc1);
  }
  // phase 2: (exp(Acum[l]) * C) @ prev^T
  const float rf0 = __expf(ac0), rf1 = __expf(ac1);
#pragma unroll
  for (int k0 = 0; k0 < NSTATE; k0 += 32) {
    v16bf cc0 = ldfragA(cm0 + k0, kh);
    v16bf cc1 = ldfragA(cm1 + k0, kh);
    v16bf a0, a1, bf;
#pragma unroll
    for (int i = 0; i < 16; ++i) {
      a0[i] = (bf16)((float)cc0[i] * rf0);
      a1[i] = (bf16)((float)cc1[i] * rf1);
      bf[i] = (bf16)pvrow[k0 + kh * 16 + i];
    }
    acc0 = wmma_bf16(a0, bf, acc0);
    acc1 = wmma_bf16(a1, bf, acc1);
  }
  float* yp = Y + ((size_t)(b * HT + h) * LSEQ + c * CHUNK) * PDIM;
#pragma unroll
  for (int rr = 0; rr < 8; ++rr) {
    int row0 = m0 + mi * 16 + kh * 8 + rr;
    yp[(size_t)row0 * PDIM + pcol]        = acc0[rr];
    yp[(size_t)(row0 + 32) * PDIM + pcol] = acc1[rr];
  }
}

// ---------------------------------------------------------------------------
// Combine 4 directions + D*x skip + SiLU(z) gate -> ycomb[b][loc][384]
// ---------------------------------------------------------------------------
__global__ void k_combine(const float* __restrict__ Y, const float* __restrict__ convo,
                          const float* __restrict__ zxbcdt, const float* __restrict__ Ds,
                          float* __restrict__ ycomb) {
  int idx = blockIdx.x * blockDim.x + threadIdx.x;   // B*L*384
  int ch = idx % DINNER;
  int bl = idx / DINNER;
  int loc = bl & (LSEQ - 1);
  int b   = bl >> 12;
  int hsub = ch >> 7, p = ch & 127;
  int hh = loc >> 6, ww = loc & 63;
  int t1 = (ww << 6) | hh;
  auto yat = [&](int h, int t) {
    return Y[((size_t)(b * HT + h) * LSEQ + t) * PDIM + p];
  };
  float y = yat(hsub, loc) + yat(3 + hsub, t1) +
            yat(6 + hsub, (LSEQ - 1) - loc) + yat(9 + hsub, (LSEQ - 1) - t1);
  float dsum = Ds[hsub] + Ds[3 + hsub] + Ds[6 + hsub] + Ds[9 + hsub];
  y += dsum * convo[((size_t)b * CONVD + ch) * LSEQ + loc];
  float zv = zxbcdt[((size_t)b * LSEQ + loc) * DIP + ch];
  y *= zv * sigmoidf_(zv);
  ycomb[idx] = y;
}

// ---------------------------------------------------------------------------
// RMS norm over 384 channels (in place), block = 128 threads per (b,loc)
// ---------------------------------------------------------------------------
__global__ void k_norm(float* __restrict__ ycomb, const float* __restrict__ norm_w) {
  __shared__ float red[128];
  float* yrow = ycomb + (size_t)blockIdx.x * DINNER;
  float s = 0.0f;
#pragma unroll
  for (int j = 0; j < 3; ++j) {
    float v = yrow[threadIdx.x + 128 * j];
    s += v * v;
  }
  red[threadIdx.x] = s;
  __syncthreads();
  for (int off = 64; off > 0; off >>= 1) {
    if (threadIdx.x < off) red[threadIdx.x] += red[threadIdx.x + off];
    __syncthreads();
  }
  float inv = rsqrtf(red[0] / (float)DINNER + 1e-5f);
#pragma unroll
  for (int j = 0; j < 3; ++j) {
    int ch = threadIdx.x + 128 * j;
    yrow[ch] = yrow[ch] * inv * norm_w[ch];
  }
}

// ---------------------------------------------------------------------------
extern "C" void kernel_launch(void* const* d_in, const int* in_sizes, int n_in,
                              void* d_out, int out_size, void* d_ws, size_t ws_size,
                              hipStream_t stream) {
  (void)in_sizes; (void)n_in; (void)out_size; (void)ws_size;
  const float* x          = (const float*)d_in[0];
  const float* in_proj_w  = (const float*)d_in[1];
  const float* conv_w     = (const float*)d_in[2];
  const float* conv_b     = (const float*)d_in[3];
  const float* dt_bias    = (const float*)d_in[4];
  const float* A_log      = (const float*)d_in[5];
  const float* Ds         = (const float*)d_in[6];
  const float* norm_w     = (const float*)d_in[7];
  const float* out_proj_w = (const float*)d_in[8];
  float* out = (float*)d_out;

  char* w = (char*)d_ws;
  auto alloc = [&](size_t bytes) -> void* {
    void* p = (void*)w;
    w += (bytes + 255) & ~(size_t)255;
    return p;
  };
  float* zxbcdt = (float*)alloc((size_t)BATCH * LSEQ * DIP * 4);
  float* convo  = (float*)alloc((size_t)BATCH * CONVD * LSEQ * 4);
  float* dts    = (float*)alloc((size_t)BATCH * LSEQ * NH3 * 4);
  float* Acum   = (float*)alloc((size_t)BATCH * NH3 * LSEQ * 4);
  float* Alast  = (float*)alloc((size_t)BATCH * NH3 * NCHUNK * 4);
  bf16*  Bmat   = (bf16*)alloc((size_t)BATCH * LSEQ * NSTATE * 2);
  bf16*  BmatT  = (bf16*)alloc((size_t)BATCH * NSTATE * LSEQ * 2);
  bf16*  Cmat   = (bf16*)alloc((size_t)BATCH * LSEQ * NSTATE * 2);
  bf16*  XdT    = (bf16*)alloc((size_t)BATCH * HT * PDIM * LSEQ * 2);
  bf16*  CB     = (bf16*)alloc((size_t)BATCH * NCHUNK * CHUNK * CHUNK * 2);
  float* states = (float*)alloc((size_t)BATCH * NCHUNK * HT * PDIM * NSTATE * 4);
  float* Yb     = (float*)alloc((size_t)BATCH * HT * LSEQ * PDIM * 4);
  float* ycomb  = (float*)alloc((size_t)BATCH * LSEQ * DINNER * 4);

  const int MT = BATCH * LSEQ;   // 16384

  // 1) in_proj GEMM: zxbcdt = x @ in_proj_w^T
  k_gemm<16384, DIP, DMODEL><<<dim3((DIP + 63) / 64, MT / 64), 256, 0, stream>>>(
      x, in_proj_w, zxbcdt);
  // 2) dt softplus
  k_dts<<<MT / 256, 256, 0, stream>>>(zxbcdt, dt_bias, dts);
  // 3) depthwise conv + SiLU
  k_conv<<<(BATCH * CONVD * LSEQ) / 256, 256, 0, stream>>>(zxbcdt, conv_w, conv_b, convo);
  // 4) per-chunk dA cumsum
  k_cumsum<<<1, BATCH * NH3 * NCHUNK, 0, stream>>>(dts, A_log, Acum, Alast);
  // 5) gathers
  k_gatherBC<<<(BATCH * LSEQ * NSTATE) / 256, 256, 0, stream>>>(convo, Bmat, BmatT, Cmat);
  k_gatherX<<<(BATCH * HT * PDIM * LSEQ) / 256, 256, 0, stream>>>(convo, dts, XdT);
  // 6) CB = C @ B^T per (b, chunk)
  k_cb<<<dim3(4, 4, BATCH * NCHUNK), 256, 0, stream>>>(Cmat, Bmat, CB);
  // 7) per-chunk end states
  k_states<<<dim3(4, 2, BATCH * NCHUNK * HT), 256, 0, stream>>>(XdT, BmatT, Acum, Alast,
                                                                states);
  // 8) inter-chunk recurrence (states -> prev, in place)
  k_recur<<<(BATCH * HT * PDIM * NSTATE) / 256, 256, 0, stream>>>(states, Alast);
  // 9) Y = Y_diag + Y_off
  k_y<<<dim3(2, 4, BATCH * NCHUNK * HT), 256, 0, stream>>>(CB, XdT, Cmat, states, Acum, Yb);
  // 10) direction combine + gate
  k_combine<<<(BATCH * LSEQ * DINNER) / 256, 256, 0, stream>>>(Yb, convo, zxbcdt, Ds, ycomb);
  // 11) RMS norm
  k_norm<<<MT, 128, 0, stream>>>(ycomb, norm_w);
  // 12) out_proj GEMM: out = ycomb @ out_proj_w^T
  k_gemm<16384, DMODEL, DINNER><<<dim3(DMODEL / 64, MT / 64), 256, 0, stream>>>(
      ycomb, out_proj_w, out);
}
